// Block_Head_83047487635490
// MI455X (gfx1250) — matrix operand order repaired and checked by reading, not compiled, not run on hardware
//
#include <hip/hip_runtime.h>

typedef unsigned short u16;
typedef unsigned int   u32;

typedef __attribute__((ext_vector_type(16))) __bf16 bf16x16;
typedef __attribute__((ext_vector_type(8)))  float  v8f;

union BF16Frag {
    bf16x16 v;
    u16     h[16];
    uint4   q[2];
};

__device__ __forceinline__ u16 f2bf(float f) {
    u32 x = __builtin_bit_cast(u32, f);
    u32 r = x + 0x7FFFu + ((x >> 16) & 1u);   // round-to-nearest-even
    return (u16)(r >> 16);
}

__device__ __forceinline__ float siluf(float x) {
    return x / (1.0f + __expf(-x));
}

// generic f32 -> bf16 cast (weight pre-conversion, once per launch)
__global__ __launch_bounds__(256)
void cast_bf16(const float* __restrict__ src, u16* __restrict__ dst, int n)
{
    int i = blockIdx.x * 256 + threadIdx.x;
    if (i < n) dst[i] = f2bf(src[i]);
}

// ---------------------------------------------------------------------------
// GEMM: C[m,n] = sum_k A[m,k] * W[n,k]
//   A: bf16 (as u16), row-major M x K        (activations)
//   W: bf16 (as u16), row-major N x K        (pre-converted weights)
//   C: f32, row-major M x N
// Requirements: M % 256 == 0, K % 32 == 0. N guarded.
// Block: 256 threads = 8 wave32s; block tile 256x128; wave tile 64x64
// (16 WMMAs / wave / 32-K step). Double-buffered LDS pipeline: global loads
// for tile k+1 issue before the WMMAs of tile k; one barrier per K-step.
// Staging is a pure packed copy (no conversion VALU, minimal live regs).
// ---------------------------------------------------------------------------
#define TS   40                  // padded LDS row stride in halves (80B)
#define ALDS (256 * TS)          // one A buffer, halves
#define BLDS (128 * TS)          // one B buffer, halves

__global__ __launch_bounds__(256)
void gemm_bf16_wmma(const u16* __restrict__ A, const u16* __restrict__ W,
                    float* __restrict__ C, int M, int N, int K)
{
    __shared__ u16 As[2 * ALDS];
    __shared__ u16 Bs[2 * BLDS];

    const int tid   = threadIdx.x;
    const int lane  = tid & 31;
    const int wv    = tid >> 5;
    const int mwave = wv & 3;        // 4 waves along M (64 rows each)
    const int nwave = wv >> 2;       // 2 waves along N (64 cols each)
    const int m0  = blockIdx.y * 256;
    const int bn0 = blockIdx.x * 128;

    const v8f vzero = {0.f,0.f,0.f,0.f,0.f,0.f,0.f,0.f};
    v8f acc[4][4];
#pragma unroll
    for (int mi = 0; mi < 4; mi++)
#pragma unroll
        for (int ni = 0; ni < 4; ni++) acc[mi][ni] = vzero;

    // staging: A -> thread stages full row tid (32 halves);
    //          B -> thread stages col (tid>>1), K-half (tid&1) (16 halves)
    const int bcol  = tid >> 1;
    const int bhalf = tid & 1;
    const int gcol  = bn0 + bcol;

    uint4 ra0, ra1, ra2, ra3;     // A row chunk (32 halves)
    uint4 rb0, rb1;               // B chunk (16 halves)

    auto load_tile = [&](int k0) {
        const uint4* gp = (const uint4*)(A + (size_t)(m0 + tid) * K + k0);
        ra0 = gp[0]; ra1 = gp[1]; ra2 = gp[2]; ra3 = gp[3];
        if (gcol < N) {
            const uint4* wp =
                (const uint4*)(W + (size_t)gcol * K + k0 + bhalf * 16);
            rb0 = wp[0]; rb1 = wp[1];
        } else {
            rb0 = make_uint4(0u, 0u, 0u, 0u);
            rb1 = make_uint4(0u, 0u, 0u, 0u);
        }
    };
    auto store_tile = [&](int buf) {
        u16* Ad = As + buf * ALDS + tid * TS;
        *(uint4*)(Ad)      = ra0;
        *(uint4*)(Ad + 8)  = ra1;
        *(uint4*)(Ad + 16) = ra2;
        *(uint4*)(Ad + 24) = ra3;
        u16* Bd = Bs + buf * BLDS + bcol * TS + bhalf * 16;
        *(uint4*)(Bd)     = rb0;
        *(uint4*)(Bd + 8) = rb1;
    };

    // prologue: stage tile 0
    load_tile(0);
    store_tile(0);
    __syncthreads();

    int p = 0;
    for (int k0 = 0; k0 < K; k0 += 32) {
        const bool has_next = (k0 + 32) < K;
        if (has_next) load_tile(k0 + 32);   // global loads overlap WMMAs below

        // ---- fragments + 16 WMMAs on buffer p ----
        // A (16-bit 16x32): lane<16 -> M=lane,    K = 0..7 & 16..23
        //                   lane>=16 -> M=lane-16, K = 8..15 & 24..31
        // B (16-bit 32x16): lane<16 -> N=lane, K = 0..7 & 16..23; lane>=16 -> K = 8..15 & 24..31
        const int kb2 = (lane >> 4) * 8;
        const int kbB = (lane >> 4) * 8;
        const u16* Ab = As + p * ALDS;
        const u16* Bb = Bs + p * BLDS;
#pragma unroll
        for (int mh = 0; mh < 2; mh++) {        // 2 M-halves: 2 A-frags live
            BF16Frag af[2];
#pragma unroll
            for (int mi = 0; mi < 2; mi++) {
                int r = mwave * 64 + (mh * 2 + mi) * 16 + (lane & 15);
                af[mi].q[0] = *(const uint4*)(Ab + r * TS + kb2);
                af[mi].q[1] = *(const uint4*)(Ab + r * TS + kb2 + 16);
            }
#pragma unroll
            for (int ni = 0; ni < 4; ni++) {
                int c = nwave * 64 + ni * 16 + (lane & 15);
                BF16Frag bf;
                bf.q[0] = *(const uint4*)(Bb + c * TS + kbB);
                bf.q[1] = *(const uint4*)(Bb + c * TS + kbB + 16);
#pragma unroll
                for (int mi = 0; mi < 2; mi++)
                    acc[mh * 2 + mi][ni] = __builtin_amdgcn_wmma_f32_16x16x32_bf16(
                        false, af[mi].v, false, bf.v, (short)0,
                        acc[mh * 2 + mi][ni], false, false);
            }
        }

        if (has_next) {
            store_tile(1 - p);
            __syncthreads();
            p ^= 1;
        }
    }

    // ---- store C (f32 16x16 layout: lane<16 -> M=r, lane>=16 -> M=8+r) ----
#pragma unroll
    for (int mi = 0; mi < 4; mi++) {
#pragma unroll
        for (int ni = 0; ni < 4; ni++) {
            int growb = m0 + mwave * 64 + mi * 16 + ((lane >> 4) * 8);
            int gc    = bn0 + nwave * 64 + ni * 16 + (lane & 15);
            if (gc < N) {
#pragma unroll
                for (int r = 0; r < 8; r++)
                    C[(size_t)(growb + r) * N + gc] = acc[mi][ni][r];
            }
        }
    }
}

// ---------------------------------------------------------------------------
// RMSNorm over D=512, write bf16 (xn). One block per token row.
// Token row -> state position: b = row/512, pos = pos_base + row%512.
// Weight selected by (rel >> wshift) & 3 (wshift=9 -> always w0, wshift=7 ->
// text segment select).
// ---------------------------------------------------------------------------
__global__ __launch_bounds__(256)
void rmsnorm_cast(const float* __restrict__ st,
                  const float* __restrict__ w0, const float* __restrict__ w1,
                  const float* __restrict__ w2, const float* __restrict__ w3,
                  int wshift, int pos_base, u16* __restrict__ out)
{
    __shared__ float red[256];
    int row = blockIdx.x;                 // 0..1023
    int b = row >> 9, rel = row & 511;
    const float* src = st + (size_t)(b * 2048 + pos_base + rel) * 512;
    int tid = threadIdx.x;
    float v0 = src[tid], v1 = src[tid + 256];
    red[tid] = v0 * v0 + v1 * v1;
    __syncthreads();
    for (int s = 128; s > 0; s >>= 1) {
        if (tid < s) red[tid] += red[tid + s];
        __syncthreads();
    }
    float scale = rsqrtf(red[0] * (1.0f / 512.0f) + 1e-6f);
    int wi = (rel >> wshift) & 3;
    const float* wsel = (wi == 0) ? w0 : (wi == 1) ? w1 : (wi == 2) ? w2 : w3;
    out[(size_t)row * 512 + tid]       = f2bf(v0 * scale * wsel[tid]);
    out[(size_t)row * 512 + tid + 256] = f2bf(v1 * scale * wsel[tid + 256]);
}

// ---------------------------------------------------------------------------
// Depthwise causal conv (width 4) + bias + silu on xi half of xz.
// xz: 1024 x 4096 (xi = cols 0..2047). Outputs xi in f32 and bf16.
// ---------------------------------------------------------------------------
__global__ __launch_bounds__(256)
void conv_silu(const float* __restrict__ xz, const float* __restrict__ cw,
               const float* __restrict__ cb, float* __restrict__ xif,
               u16* __restrict__ xib)
{
    int idx = blockIdx.x * 256 + threadIdx.x;   // < 1024*2048
    int t = idx >> 11, d = idx & 2047;
    int b = t >> 9, l = t & 511;
    float acc = cb[d];
#pragma unroll
    for (int j = 0; j < 4; j++) {
        int li = l - 3 + j;
        if (li >= 0)
            acc += xz[(size_t)(b * 512 + li) * 4096 + d] * cw[d * 4 + j];
    }
    float v = siluf(acc);
    xif[idx] = v;
    xib[idx] = f2bf(v);
}

// dt columns (0..31) of x_db -> bf16 for dt_proj GEMM
__global__ __launch_bounds__(256)
void dt_cast(const float* __restrict__ xdb, u16* __restrict__ dtb)
{
    int idx = blockIdx.x * 256 + threadIdx.x;   // < 1024*32
    int row = idx >> 5, c = idx & 31;
    dtb[idx] = f2bf(xdb[(size_t)row * 288 + c]);
}

// delta = softplus(x + dt_bias[d]), in place on GEMM output (1024 x 2048)
__global__ __launch_bounds__(256)
void softplus_bias(float* __restrict__ g, const float* __restrict__ bias)
{
    int idx = blockIdx.x * 256 + threadIdx.x;   // < 1024*2048
    int d = idx & 2047;
    float x = g[idx] + bias[d];
    g[idx] = (x > 20.0f) ? x : log1pf(__expf(x));
}

// ---------------------------------------------------------------------------
// Selective scan. One wave per (b,d) channel; 4 states per lane (128 states).
// y[t,d] = sum_n h_n * C_n;  h_n <- h_n*exp(delta*A_n) + delta*u*B_n
// Epilogue: (y + u*D[d]) * silu(z), emit bf16 for out_proj GEMM.
// ---------------------------------------------------------------------------
__global__ __launch_bounds__(256)
void ssm_scan(const float* __restrict__ delta, const float* __restrict__ xif,
              const float* __restrict__ xdb, const float* __restrict__ xz,
              const float* __restrict__ A_log, const float* __restrict__ Dv,
              u16* __restrict__ ybf)
{
    int lane = threadIdx.x & 31;
    int wv   = threadIdx.x >> 5;
    int ch   = blockIdx.x * 8 + wv;             // 0..4095
    int b = ch >> 11, d = ch & 2047;
    int n0 = lane * 4;
    const float* al = A_log + (size_t)d * 128 + n0;
    float a0 = -__expf(al[0]), a1 = -__expf(al[1]);
    float a2 = -__expf(al[2]), a3 = -__expf(al[3]);
    float h0 = 0.f, h1 = 0.f, h2 = 0.f, h3 = 0.f;
    float Dd = Dv[d];
    for (int l = 0; l < 512; l++) {
        size_t row = (size_t)(b * 512 + l);
        float dt = delta[row * 2048 + d];
        float u  = xif[row * 2048 + d];
        float4 Bv = *(const float4*)(xdb + row * 288 + 32 + n0);
        float4 Cv = *(const float4*)(xdb + row * 288 + 160 + n0);
        float du = dt * u;
        h0 = h0 * __expf(dt * a0) + du * Bv.x;
        h1 = h1 * __expf(dt * a1) + du * Bv.y;
        h2 = h2 * __expf(dt * a2) + du * Bv.z;
        h3 = h3 * __expf(dt * a3) + du * Bv.w;
        float y = h0 * Cv.x + h1 * Cv.y + h2 * Cv.z + h3 * Cv.w;
        y += __shfl_xor(y, 16);
        y += __shfl_xor(y, 8);
        y += __shfl_xor(y, 4);
        y += __shfl_xor(y, 2);
        y += __shfl_xor(y, 1);
        if (lane == 0) {
            float z = xz[row * 4096 + 2048 + d];
            ybf[row * 2048 + d] = f2bf((y + u * Dd) * siluf(z));
        }
    }
}

// h = (y + b1[c]) * silu(gate + b1[H+c]); fc1 output is 1024 x 2H
__global__ __launch_bounds__(256)
void gate_silu(const float* __restrict__ y, const float* __restrict__ b1,
               u16* __restrict__ hb, int H)
{
    int idx = blockIdx.x * 256 + threadIdx.x;   // < 1024*H
    int row = idx / H, c = idx % H;
    float a = y[(size_t)row * 2 * H + c] + b1[c];
    float g = y[(size_t)row * 2 * H + H + c] + b1[H + c];
    hb[idx] = f2bf(a * siluf(g));
}

// out[b, pos_base+rel, c] += src[row,c] (+ bias[c])
__global__ __launch_bounds__(256)
void add_resid(float* __restrict__ out, const float* __restrict__ src,
               const float* __restrict__ bias, int pos_base)
{
    int idx = blockIdx.x * 256 + threadIdx.x;   // < 1024*512
    int row = idx >> 9, c = idx & 511;
    int b = row >> 9, rel = row & 511;
    float v = src[idx] + (bias ? bias[c] : 0.0f);
    out[(size_t)(b * 2048 + pos_base + rel) * 512 + c] += v;
}

// ---------------------------------------------------------------------------
// Host side
// ---------------------------------------------------------------------------
struct MP { const float *inp, *cw, *cb, *xp, *dtw, *dtbias, *alog, *dvec, *outp; };
struct GP { const float *w1, *b1, *w2, *b2; };

extern "C" void kernel_launch(void* const* d_in, const int* in_sizes, int n_in,
                              void* d_out, int out_size, void* d_ws, size_t ws_size,
                              hipStream_t stream)
{
    (void)in_sizes; (void)n_in; (void)out_size; (void)ws_size;
    const float* x = (const float*)d_in[0];
    // indices: 0 x, 1 T_motion, 2 text_meta, 3..12 norms,
    // 13..21 mamba_music, 22..30 mamba_up, 31..39 mamba_down,
    // 40..43/44..47/48..51 gate_mlp_1..3, 52..55 text_mlp
    const float* nmw[10];
    for (int i = 0; i < 10; i++) nmw[i] = (const float*)d_in[3 + i];

    MP mp[3];
    const int mbase[3] = {13, 22, 31};
    for (int i = 0; i < 3; i++) {
        int b = mbase[i];
        mp[i].inp    = (const float*)d_in[b + 0];
        mp[i].cw     = (const float*)d_in[b + 1];
        mp[i].cb     = (const float*)d_in[b + 2];
        mp[i].xp     = (const float*)d_in[b + 3];
        mp[i].dtw    = (const float*)d_in[b + 4];
        mp[i].dtbias = (const float*)d_in[b + 5];
        mp[i].alog   = (const float*)d_in[b + 6];
        mp[i].dvec   = (const float*)d_in[b + 7];
        mp[i].outp   = (const float*)d_in[b + 8];
    }
    GP gp[3], tp;
    const int gbase[3] = {40, 44, 48};
    for (int i = 0; i < 3; i++) {
        int b = gbase[i];
        gp[i].w1 = (const float*)d_in[b + 0];
        gp[i].b1 = (const float*)d_in[b + 1];
        gp[i].w2 = (const float*)d_in[b + 2];
        gp[i].b2 = (const float*)d_in[b + 3];
    }
    tp.w1 = (const float*)d_in[52]; tp.b1 = (const float*)d_in[53];
    tp.w2 = (const float*)d_in[54]; tp.b2 = (const float*)d_in[55];

    float* out = (float*)d_out;

    // workspace carve
    char* p = (char*)d_ws;
    auto carve = [&](size_t bytes) {
        char* r = p;
        p += (bytes + 255) & ~(size_t)255;
        return r;
    };
    // activations
    u16*   xn  = (u16*)  carve((size_t)1024 * 512  * 2);  // normed bf16 input
    float* xz  = (float*)carve((size_t)1024 * 4096 * 4);  // in_proj out
    float* xif = (float*)carve((size_t)1024 * 2048 * 4);  // conv+silu f32
    u16*   xib = (u16*)  carve((size_t)1024 * 2048 * 2);  // conv+silu bf16
    float* xdb = (float*)carve((size_t)1024 * 288  * 4);  // x_proj out
    u16*   dtb = (u16*)  carve((size_t)1024 * 32   * 2);  // dt bf16
    float* go  = (float*)carve((size_t)1024 * 4096 * 4);  // gemm scratch / delta
    u16*   ybf = (u16*)  carve((size_t)1024 * 2048 * 2);  // scan out bf16
    float* mo  = (float*)carve((size_t)1024 * 512  * 4);  // branch output
    u16*   hb  = (u16*)  carve((size_t)1024 * 2048 * 2);  // gated hidden bf16
    // bf16 weight cache (pre-converted once per launch, L2-resident after)
    u16* w_inp[3]; u16* w_xp[3]; u16* w_dtw[3]; u16* w_outp[3];
    u16* w_g1[3];  u16* w_g2[3];
    for (int i = 0; i < 3; i++) {
        w_inp[i]  = (u16*)carve((size_t)4096 * 512  * 2);
        w_xp[i]   = (u16*)carve((size_t)288  * 2048 * 2);
        w_dtw[i]  = (u16*)carve((size_t)2048 * 32   * 2);
        w_outp[i] = (u16*)carve((size_t)512  * 2048 * 2);
        w_g1[i]   = (u16*)carve((size_t)1536 * 512  * 2);
        w_g2[i]   = (u16*)carve((size_t)512  * 768  * 2);
    }
    u16* w_t1 = (u16*)carve((size_t)4096 * 512  * 2);
    u16* w_t2 = (u16*)carve((size_t)512  * 2048 * 2);

    auto cast = [&](const float* src, u16* dst, int n) {
        cast_bf16<<<(n + 255) / 256, 256, 0, stream>>>(src, dst, n);
    };
    auto gemm = [&](const u16* A, const u16* W, float* C, int M, int N, int K) {
        dim3 grid((N + 127) / 128, M / 256);
        gemm_bf16_wmma<<<grid, 256, 0, stream>>>(A, W, C, M, N, K);
    };

    // ---- weight pre-conversion (f32 -> bf16) ----
    for (int i = 0; i < 3; i++) {
        cast(mp[i].inp,  w_inp[i],  4096 * 512);
        cast(mp[i].xp,   w_xp[i],   288 * 2048);
        cast(mp[i].dtw,  w_dtw[i],  2048 * 32);
        cast(mp[i].outp, w_outp[i], 512 * 2048);
        cast(gp[i].w1,   w_g1[i],   1536 * 512);
        cast(gp[i].w2,   w_g2[i],   512 * 768);
    }
    cast(tp.w1, w_t1, 4096 * 512);
    cast(tp.w2, w_t2, 512 * 2048);

    // residual base = x
    hipMemcpyAsync(out, x, (size_t)2 * 2048 * 512 * sizeof(float),
                   hipMemcpyDeviceToDevice, stream);

    // ---- motion segments: music / up / down ----
    for (int s = 0; s < 3; s++) {
        const float* nw1 = nmw[2 * s + 0];
        const float* nw2 = nmw[2 * s + 1];
        int pos = s * 512;
        // mamba branch
        rmsnorm_cast<<<1024, 256, 0, stream>>>(out, nw1, nw1, nw1, nw1, 9, pos, xn);
        gemm(xn, w_inp[s], xz, 1024, 4096, 512);
        conv_silu<<<8192, 256, 0, stream>>>(xz, mp[s].cw, mp[s].cb, xif, xib);
        gemm(xib, w_xp[s], xdb, 1024, 288, 2048);
        dt_cast<<<128, 256, 0, stream>>>(xdb, dtb);
        gemm(dtb, w_dtw[s], go, 1024, 2048, 32);
        softplus_bias<<<8192, 256, 0, stream>>>(go, mp[s].dtbias);
        ssm_scan<<<512, 256, 0, stream>>>(go, xif, xdb, xz, mp[s].alog, mp[s].dvec, ybf);
        gemm(ybf, w_outp[s], mo, 1024, 512, 2048);
        add_resid<<<2048, 256, 0, stream>>>(out, mo, nullptr, pos);
        // gated MLP branch (H = 768)
        rmsnorm_cast<<<1024, 256, 0, stream>>>(out, nw2, nw2, nw2, nw2, 9, pos, xn);
        gemm(xn, w_g1[s], go, 1024, 1536, 512);
        gate_silu<<<3072, 256, 0, stream>>>(go, gp[s].b1, hb, 768);
        gemm(hb, w_g2[s], mo, 1024, 512, 768);
        add_resid<<<2048, 256, 0, stream>>>(out, mo, gp[s].b2, pos);
    }

    // ---- text segments (4 x 128 tokens, shared text MLP, H = 2048) ----
    rmsnorm_cast<<<1024, 256, 0, stream>>>(out, nmw[6], nmw[7], nmw[8], nmw[9],
                                           7, 1536, xn);
    gemm(xn, w_t1, go, 1024, 4096, 512);
    gate_silu<<<8192, 256, 0, stream>>>(go, tp.b1, hb, 2048);
    gemm(hb, w_t2, mo, 1024, 512, 2048);
    add_resid<<<2048, 256, 0, stream>>>(out, mo, tp.b2, 1536);
}
